// DGCNN_87119116632531
// MI455X (gfx1250) — matrix operand order repaired and checked by reading, not compiled, not run on hardware
//
#include <hip/hip_runtime.h>
#include <hip/hip_bf16.h>

// ---------------- problem constants (from reference) ----------------
#define N_NODES 131072
#define N_EDGES 1048576
#define NGRAPH  1024
#define HDIM    128
#define DCAT    384            // 3 * HDIM
#define TOPK    30
#define C1CH    16
#define C2CH    32
#define KERN5   5
#define POOLL   13             // (30-5+1)/2
#define FLAT    416            // 32*13
#define EPSBN   1e-5f

typedef __attribute__((ext_vector_type(16))) _Float16 v16h;
typedef __attribute__((ext_vector_type(8)))  _Float16 v8h;
typedef __attribute__((ext_vector_type(8)))  float    v8f;

// ---------------- init: deg=1 (self loop), counts/starts/topk ----------------
__global__ void k_init(float* deg, int* counts, int* starts, int* topk) {
    int i = blockIdx.x * blockDim.x + threadIdx.x;
    if (i < N_NODES) deg[i] = 1.0f;
    if (i < NGRAPH) { counts[i] = 0; starts[i] = N_NODES; }
    if (i < NGRAPH * TOPK) topk[i] = -1;
}

// ---------------- degree over target nodes ----------------
__global__ void k_degree(const int* __restrict__ ei, float* __restrict__ deg) {
    int e = blockIdx.x * blockDim.x + threadIdx.x;
    if (e < N_EDGES) atomicAdd(deg + ei[N_EDGES + e], 1.0f);
}

__global__ void k_dis(const float* __restrict__ deg, float* __restrict__ dis) {
    int i = blockIdx.x * blockDim.x + threadIdx.x;
    if (i < N_NODES) dis[i] = rsqrtf(deg[i]);
}

// ---------------- convert GCN weights to f16 (WMMA operand precision) --------
__global__ void k_cvt(const float* __restrict__ W1, const float* __restrict__ W2,
                      _Float16* __restrict__ W1h, _Float16* __restrict__ W2h) {
    int i = blockIdx.x * blockDim.x + threadIdx.x;   // < 128*128
    W1h[i] = (_Float16)W1[i];
    W2h[i] = (_Float16)W2[i];
}

// ---------------- layer 0: one_hot(z) @ W0 == row gather ----------------
__global__ void k_gather0(const int* __restrict__ z, const float* __restrict__ W0,
                          float* __restrict__ h) {
    int t = blockIdx.x * blockDim.x + threadIdx.x;   // N*128
    int i = t >> 7, f = t & 127;
    h[t] = W0[z[i] * HDIM + f];
}

// ---------------- acc = dis[i]^2 * h[i] + b  (self-loop + bias init) ---------
__global__ void k_selfbias(const float* __restrict__ dis, const float* __restrict__ h,
                           const float* __restrict__ bvec, float* __restrict__ acc) {
    int t = blockIdx.x * blockDim.x + threadIdx.x;   // N*128
    int i = t >> 7, f = t & 127;
    float d = dis[i];
    acc[t] = d * d * h[t] + bvec[f];
}

// ---------------- edge scatter: acc[col] += dis[row]*dis[col]*h[row] ---------
// thread handles 4 features of one edge (float4 gather, 4 f32 L2 atomics)
__global__ void k_scatter(const int* __restrict__ ei, const float* __restrict__ dis,
                          const float* __restrict__ h, float* __restrict__ acc) {
    int t = blockIdx.x * blockDim.x + threadIdx.x;   // E*32
    int e = t >> 5;
    int f4 = (t & 31) << 2;
    int r = ei[e], c = ei[N_EDGES + e];
    float nrm = dis[r] * dis[c];
    const float4 hv = *reinterpret_cast<const float4*>(h + (size_t)r * HDIM + f4);
    float* a = acc + (size_t)c * HDIM + f4;
    atomicAdd(a + 0, nrm * hv.x);
    atomicAdd(a + 1, nrm * hv.y);
    atomicAdd(a + 2, nrm * hv.z);
    atomicAdd(a + 3, nrm * hv.w);
}

// ---------------- tanh, write f16 slice of xcat (also last-channel key) ------
__global__ void k_tanh(const float* __restrict__ acc, _Float16* __restrict__ xcat,
                       float* __restrict__ key, int off, int wkey) {
    int t = blockIdx.x * blockDim.x + threadIdx.x;   // N*128
    int i = t >> 7, f = t & 127;
    float v = tanhf(acc[t]);
    xcat[(size_t)i * DCAT + off + f] = (_Float16)v;
    if (wkey && f == HDIM - 1) key[i] = v;
}

// ---------------- WMMA GEMM: Hout[N,128] = X[:, off:off+128] @ Wh[128,128] ---
// Block = 8 waves; wave w owns a 16-col strip, keeps its B fragments
// (32KB weight matrix) in registers across 16 row tiles. f16 in, f32 acc.
// A/B use contiguous k = 16*half + j; C/D layout per CDNA5 ISA §7.12.2.
__global__ __launch_bounds__(256) void k_gemm(const _Float16* __restrict__ X, int inOff,
                                              const _Float16* __restrict__ Wh,
                                              float* __restrict__ Hout) {
    const int lane = threadIdx.x & 31;
    const int wave = threadIdx.x >> 5;
    const int colBase = wave * 16;
    const int half = lane >> 4;     // lane group
    const int m = lane & 15;

    // B 32x16 f16: lane n<16 -> col colBase+n, K = kB..kB+15 ; lane n+16 -> K+16..K+31
    v16h bf[4];
#pragma unroll
    for (int s = 0; s < 4; ++s) {
        int kB = 32 * s + (half << 4);
#pragma unroll
        for (int j = 0; j < 16; ++j)
            bf[s][j] = Wh[(size_t)(kB + j) * HDIM + colBase + m];
    }

    const int rowsStart = blockIdx.x * 256;
    for (int rt = 0; rt < 16; ++rt) {
        int rowBase = rowsStart + rt * 16;
        const _Float16* arow = X + (size_t)(rowBase + m) * DCAT + inOff;
        if (rt + 1 < 16)  // stream next A tile through GL2 (global_prefetch_b8)
            __builtin_prefetch(X + (size_t)(rowBase + 16 + m) * DCAT + inOff, 0, 1);

        v8f c = {};
#pragma unroll
        for (int s = 0; s < 4; ++s) {
            // A 16x32 f16: lane m<16 -> row m, K 0..15; lane m+16 -> K 16..31
            const _Float16* pa = arow + 32 * s + (half << 4);
            v8h lo = *reinterpret_cast<const v8h*>(pa);
            v8h hi = *reinterpret_cast<const v8h*>(pa + 8);
            v16h a;
#pragma unroll
            for (int j = 0; j < 8; ++j) { a[j] = lo[j]; a[8 + j] = hi[j]; }
            c = __builtin_amdgcn_wmma_f32_16x16x32_f16(false, a, false, bf[s],
                                                       (short)0, c, false, false);
        }
        // C/D f32 16x16: VGPR r -> (M = r + 8*half, N = colBase + m)
        int mBase = rowBase + (half << 3);
        int n = colBase + m;
#pragma unroll
        for (int r = 0; r < 8; ++r)
            Hout[(size_t)(mBase + r) * HDIM + n] = c[r];
    }
}

// ---------------- per-graph counts / starts (batch is sorted) ----------------
__global__ void k_counts(const int* __restrict__ batch, int* __restrict__ counts,
                         int* __restrict__ starts) {
    int i = blockIdx.x * blockDim.x + threadIdx.x;
    if (i < N_NODES) {
        int b = batch[i];
        atomicAdd(counts + b, 1);
        atomicMin(starts + b, i);
    }
}

// ---------------- stable top-K by descending key (rank-by-counting) ----------
__global__ void k_topk(const float* __restrict__ key, const int* __restrict__ counts,
                       const int* __restrict__ starts, int* __restrict__ topk) {
    int b = blockIdx.x;
    int cnt = counts[b];
    int s = starts[b];
    if (cnt <= 0) return;
    for (int p = threadIdx.x; p < cnt; p += blockDim.x) {
        int i = s + p;
        float ki = key[i];
        int rank = 0;
        for (int q = 0; q < cnt; ++q) {
            int j = s + q;
            float kj = key[j];
            rank += (kj > ki) || (kj == ki && j < i);
        }
        if (rank < TOPK) topk[b * TOPK + rank] = i;
    }
}

// ---------------- conv1 (== dense over D per kept node) + relu + BN ----------
__global__ void k_conv1(const _Float16* __restrict__ xcat, const int* __restrict__ topk,
                        const float* __restrict__ Wc1, const float* __restrict__ bc1,
                        const float* __restrict__ g1, const float* __restrict__ be1,
                        const float* __restrict__ rm1, const float* __restrict__ rv1,
                        float* __restrict__ h1) {
    int b = blockIdx.x;
    int t = threadIdx.x;            // 0..479
    int c = t / TOPK;
    int k = t - c * TOPK;
    int node = topk[b * TOPK + k];
    float v = bc1[c];
    if (node >= 0) {
        const _Float16* xr = xcat + (size_t)node * DCAT;
        const float* w = Wc1 + c * DCAT;
        float acc = 0.f;
#pragma unroll 4
        for (int d = 0; d < DCAT; ++d) acc += (float)xr[d] * w[d];
        v += acc;
    }
    v = fmaxf(v, 0.f);
    v = (v - rm1[c]) * (g1[c] * rsqrtf(rv1[c] + EPSBN)) + be1[c];
    h1[((size_t)b * C1CH + c) * TOPK + k] = v;
}

// ---------------- conv2 + relu + BN + maxpool(2,2) ----------------
__global__ void k_conv2pool(const float* __restrict__ h1, const float* __restrict__ Wc2,
                            const float* __restrict__ bc2, const float* __restrict__ g2,
                            const float* __restrict__ be2, const float* __restrict__ rm2,
                            const float* __restrict__ rv2, float* __restrict__ hp) {
    __shared__ float sh[C1CH * TOPK];
    int b = blockIdx.x;
    for (int q = threadIdx.x; q < C1CH * TOPK; q += blockDim.x)
        sh[q] = h1[(size_t)b * C1CH * TOPK + q];
    __syncthreads();
    int o = threadIdx.x / POOLL;    // 0..31
    int u = threadIdx.x - o * POOLL;
    float scale = g2[o] * rsqrtf(rv2[o] + EPSBN);
    float best = -3.4e38f;
#pragma unroll
    for (int s = 0; s < 2; ++s) {
        int tpos = 2 * u + s;       // 0..25
        float v = bc2[o];
#pragma unroll
        for (int c = 0; c < C1CH; ++c)
#pragma unroll
            for (int j = 0; j < KERN5; ++j)
                v += sh[c * TOPK + tpos + j] * Wc2[(o * C1CH + c) * KERN5 + j];
        v = fmaxf(v, 0.f);
        v = (v - rm2[o]) * scale + be2[o];
        best = fmaxf(best, v);
    }
    hp[((size_t)b * C2CH + o) * POOLL + u] = best;
}

// ---------------- dense head: relu(Wl1 x + bl1) -> Wl2 . y + bl2 -------------
__global__ void k_head(const float* __restrict__ hp, const float* __restrict__ Wl1,
                       const float* __restrict__ bl1, const float* __restrict__ Wl2,
                       const float* __restrict__ bl2, float* __restrict__ out) {
    __shared__ float sf[FLAT];
    __shared__ float red[128];
    int b = blockIdx.x, j = threadIdx.x;
    for (int q = j; q < FLAT; q += blockDim.x) sf[q] = hp[(size_t)b * FLAT + q];
    __syncthreads();
    float acc = bl1[j];
    const float* w = Wl1 + (size_t)j * FLAT;
#pragma unroll 4
    for (int q = 0; q < FLAT; ++q) acc += sf[q] * w[q];
    acc = fmaxf(acc, 0.f);
    red[j] = acc * Wl2[j];
    __syncthreads();
    for (int s = 64; s > 0; s >>= 1) {
        if (j < s) red[j] += red[j + s];
        __syncthreads();
    }
    if (j == 0) out[b] = red[0] + bl2[0];
}

// ---------------- host driver ----------------
extern "C" void kernel_launch(void* const* d_in, const int* in_sizes, int n_in,
                              void* d_out, int out_size, void* d_ws, size_t ws_size,
                              hipStream_t stream) {
    const int*   z     = (const int*)d_in[0];
    const int*   ei    = (const int*)d_in[1];
    const int*   batch = (const int*)d_in[2];
    const float* W0    = (const float*)d_in[3];
    const float* b0    = (const float*)d_in[4];
    const float* W1    = (const float*)d_in[5];
    const float* b1    = (const float*)d_in[6];
    const float* W2    = (const float*)d_in[7];
    const float* b2    = (const float*)d_in[8];
    const float* Wc1   = (const float*)d_in[9];
    const float* bc1   = (const float*)d_in[10];
    const float* Wc2   = (const float*)d_in[11];
    const float* bc2   = (const float*)d_in[12];
    const float* g1    = (const float*)d_in[13];
    const float* be1   = (const float*)d_in[14];
    const float* g2    = (const float*)d_in[15];
    const float* be2   = (const float*)d_in[16];
    const float* Wl1   = (const float*)d_in[17];
    const float* bl1   = (const float*)d_in[18];
    const float* Wl2   = (const float*)d_in[19];
    const float* bl2   = (const float*)d_in[20];
    const float* rm1   = (const float*)d_in[21];
    const float* rv1   = (const float*)d_in[22];
    const float* rm2   = (const float*)d_in[23];
    const float* rv2   = (const float*)d_in[24];
    float* out = (float*)d_out;

    // workspace carve-up (256B aligned)
    char* base = (char*)d_ws;
    size_t off = 0;
    auto carve = [&](size_t bytes) -> char* {
        char* p = base + off;
        off = (off + bytes + 255) & ~(size_t)255;
        return p;
    };
    float*    deg    = (float*)carve((size_t)N_NODES * 4);
    float*    dis    = (float*)carve((size_t)N_NODES * 4);
    float*    key    = (float*)carve((size_t)N_NODES * 4);
    int*      counts = (int*)  carve((size_t)NGRAPH * 4);
    int*      starts = (int*)  carve((size_t)NGRAPH * 4);
    int*      topk   = (int*)  carve((size_t)NGRAPH * TOPK * 4);
    _Float16* W1h    = (_Float16*)carve((size_t)HDIM * HDIM * 2);
    _Float16* W2h    = (_Float16*)carve((size_t)HDIM * HDIM * 2);
    float*    h      = (float*)carve((size_t)N_NODES * HDIM * 4);
    float*    acc    = (float*)carve((size_t)N_NODES * HDIM * 4);
    _Float16* xcat   = (_Float16*)carve((size_t)N_NODES * DCAT * 2);
    float*    h1     = (float*)carve((size_t)NGRAPH * C1CH * TOPK * 4);
    float*    hp     = (float*)carve((size_t)NGRAPH * FLAT * 4);

    const int NF      = N_NODES * HDIM;        // 16.7M
    const dim3 B256(256);
    const int gN   = N_NODES / 256;            // 512
    const int gE   = N_EDGES / 256;            // 4096
    const int gNF  = NF / 256;                 // 65536
    const int gSC  = (N_EDGES * 32) / 256;     // 131072
    const int gGE  = N_NODES / 256;            // 512 gemm blocks (256 rows each)

    // --- normalization ---
    k_init  <<<gN,  B256, 0, stream>>>(deg, counts, starts, topk);
    k_degree<<<gE,  B256, 0, stream>>>(ei, deg);
    k_dis   <<<gN,  B256, 0, stream>>>(deg, dis);
    k_cvt   <<<(HDIM * HDIM) / 256, B256, 0, stream>>>(W1, W2, W1h, W2h);

    // --- GCN layer 0 (one-hot gather) ---
    k_gather0 <<<gNF, B256, 0, stream>>>(z, W0, h);
    k_selfbias<<<gNF, B256, 0, stream>>>(dis, h, b0, acc);
    k_scatter <<<gSC, B256, 0, stream>>>(ei, dis, h, acc);
    k_tanh    <<<gNF, B256, 0, stream>>>(acc, xcat, key, 0, 0);

    // --- GCN layer 1 (WMMA GEMM) ---
    k_gemm    <<<gGE, B256, 0, stream>>>(xcat, 0, W1h, h);
    k_selfbias<<<gNF, B256, 0, stream>>>(dis, h, b1, acc);
    k_scatter <<<gSC, B256, 0, stream>>>(ei, dis, h, acc);
    k_tanh    <<<gNF, B256, 0, stream>>>(acc, xcat, key, HDIM, 0);

    // --- GCN layer 2 (WMMA GEMM) ---
    k_gemm    <<<gGE, B256, 0, stream>>>(xcat, HDIM, W2h, h);
    k_selfbias<<<gNF, B256, 0, stream>>>(dis, h, b2, acc);
    k_scatter <<<gSC, B256, 0, stream>>>(ei, dis, h, acc);
    k_tanh    <<<gNF, B256, 0, stream>>>(acc, xcat, key, 2 * HDIM, 1);

    // --- global_sort_pool ---
    k_counts<<<gN, B256, 0, stream>>>(batch, counts, starts);
    k_topk  <<<NGRAPH, dim3(64),  0, stream>>>(key, counts, starts, topk);

    // --- conv / BN / pool / head ---
    k_conv1    <<<NGRAPH, dim3(C1CH * TOPK), 0, stream>>>(xcat, topk, Wc1, bc1,
                                                          g1, be1, rm1, rv1, h1);
    k_conv2pool<<<NGRAPH, dim3(C2CH * POOLL), 0, stream>>>(h1, Wc2, bc2, g2, be2,
                                                           rm2, rv2, hp);
    k_head     <<<NGRAPH, dim3(128), 0, stream>>>(hp, Wl1, bl1, Wl2, bl2, out);
}
